// HL_HGCNN_CIFAR10SP_dense_int3_attpool_87247965651043
// MI455X (gfx1250) — compile-verified, matrix operand order checked by reading, not run on hardware
//
#include <hip/hip_runtime.h>
#include <math.h>

// ---------------------------------------------------------------------------
// HL-HGCNN forward for MI455X (gfx1250).
// Dense GEMMs: pre-packed bf16 operands -> inner loop = 4x global_load_b128
//              + v_wmma_f32_16x16x32_bf16 (one 16x16 tile per wave32).
// Sparse scatter/gather/BN -> grid-stride VALU kernels with float atomics.
// ---------------------------------------------------------------------------

typedef __attribute__((ext_vector_type(16))) __bf16 v16bf;
typedef __attribute__((ext_vector_type(8)))  float  v8f;

union BF16x16 { v16bf v; unsigned short us[16]; uint4 q[2]; };

__device__ __forceinline__ unsigned short f2bf(float f) {
  unsigned int u = __builtin_bit_cast(unsigned int, f);
  unsigned int r = u + 0x7FFFu + ((u >> 16) & 1u);   // round to nearest even
  return (unsigned short)(r >> 16);
}

// ---- operand packing -------------------------------------------------------

#define GSL(i, n) \
  for (long long i = (long long)blockIdx.x * blockDim.x + threadIdx.x; i < (n); \
       i += (long long)gridDim.x * blockDim.x)

// A[M,K] (row stride lda, fp32) -> packed bf16 [M, Kp], zero-padded K..Kp
__global__ void k_a_pack(const float* __restrict__ A, int lda, long long M,
                         int K, int Kp, unsigned short* __restrict__ out) {
  GSL(i, M * (long long)Kp) {
    const long long r = i / Kp; const int k = (int)(i - r * Kp);
    out[i] = f2bf(k < K ? A[r * (long long)lda + k] : 0.0f);
  }
}

// W[K,N] (fp32, row-major) -> transposed packed bf16 [Np, Kp], zero-padded
__global__ void k_w_pack(const float* __restrict__ W, int N, int K, int Kp,
                         int Np, unsigned short* __restrict__ out) {
  GSL(i, (long long)Np * Kp) {
    const long long c = i / Kp; const int k = (int)(i - c * Kp);
    out[i] = f2bf((k < K && c < (long long)N) ? W[(long long)k * N + (int)c] : 0.0f);
  }
}

// ---- WMMA GEMM -------------------------------------------------------------
// C[M,N] = A @ W (+bias) (+Cin). Abf: [M,Kp] bf16, Bbf: [Np,Kp] bf16 (W^T).
// 4 waves / 128-thread block; each wave owns one 16x16 C tile.
// Fragment layout per ISA 7.12.2: per lane, elements 0..7 = K (k0+h*8 ..+7),
// elements 8..15 = K (k0+16+h*8 ..+7) -> two contiguous 16B loads each.
__global__ __launch_bounds__(128)
void gemm_bf16_wmma(const unsigned short* __restrict__ Abf,
                    const unsigned short* __restrict__ Bbf,
                    const float* __restrict__ bias,
                    const float* __restrict__ Cin,
                    float* __restrict__ C, int ldc,
                    int M, int N, int Kp) {
  const int wave = threadIdx.x >> 5;
  const int lane = threadIdx.x & 31;
  const int tile_m = (blockIdx.x * 4 + wave) * 16;
  const int tile_n = blockIdx.y * 16;
  if (tile_m >= M) return;                       // wave-uniform: EXEC stays full

  const int h   = lane >> 4;
  const int l16 = lane & 15;
  const unsigned short* __restrict__ ap = Abf + (long long)(tile_m + l16) * Kp + h * 8;
  const unsigned short* __restrict__ bp = Bbf + (long long)(tile_n + l16) * Kp + h * 8;

  v8f acc = {};
  for (int k0 = 0; k0 < Kp; k0 += 32) {
    BF16x16 a, b;
    a.q[0] = *(const uint4*)(ap);
    a.q[1] = *(const uint4*)(ap + 16);
    b.q[0] = *(const uint4*)(bp);
    b.q[1] = *(const uint4*)(bp + 16);
    ap += 32; bp += 32;
    acc = __builtin_amdgcn_wmma_f32_16x16x32_bf16(
        false, a.v, false, b.v, (short)0, acc, false, false);
  }

  const int col = tile_n + l16;
  if (col < N) {
    float badd = bias ? bias[col] : 0.0f;
#pragma unroll
    for (int r = 0; r < 8; ++r) {
      const int row = tile_m + h * 8 + r;        // ISA C/D layout
      float v = acc[r] + badd;
      if (Cin) v += Cin[(long long)row * ldc + col];
      C[(long long)row * ldc + col] = v;
    }
  }
}

// ---------------- elementwise / sparse kernels -----------------------------

__global__ void k_fill(float* __restrict__ p, float v, long long n) {
  GSL(i, n) p[i] = v;
}

__global__ void k_copy(float* __restrict__ d, const float* __restrict__ s, long long n) {
  GSL(i, n) d[i] = s[i];
}

__global__ void k_degree(const int* __restrict__ inc, long long n2, float* __restrict__ D) {
  GSL(i, n2) atomicAdd(&D[inc[i]], 1.0f);
}

// agg[u] += h_s[e], agg[v] += h_s[e]
__global__ void k_inc_scatter(const float* __restrict__ h_s, const int* __restrict__ inc,
                              long long E, int dv, float* __restrict__ agg) {
  GSL(i, E * dv) {
    const long long e = i / dv; const int f = (int)(i - e * dv);
    const float hv = h_s[i];
    atomicAdd(&agg[(long long)inc[e] * dv + f], hv);
    atomicAdd(&agg[(long long)inc[E + e] * dv + f], hv);
  }
}

// x_s = sigmoid(h_s + h_t[u]/D[u] + h_t[v]/D[v])  (in place on h_s)
__global__ void k_edge_sigmoid(float* __restrict__ h_s, const float* __restrict__ h_t,
                               const int* __restrict__ inc, const float* __restrict__ D,
                               long long E, int dv) {
  GSL(i, E * dv) {
    const long long e = i / dv; const int f = (int)(i - e * dv);
    const int u = inc[e], v = inc[E + e];
    const float x = h_s[i] + h_t[(long long)u * dv + f] / D[u]
                           + h_t[(long long)v * dv + f] / D[v];
    h_s[i] = 1.0f / (1.0f + __expf(-x));
  }
}

// x_t = sigmoid(h_t + agg/D)  (in place on h_t)
__global__ void k_node_sigmoid(float* __restrict__ h_t, const float* __restrict__ agg,
                               const float* __restrict__ D, long long N, int dv) {
  GSL(i, N * dv) {
    const long long n = i / dv;
    const float x = h_t[i] + agg[i] / D[n];
    h_t[i] = 1.0f / (1.0f + __expf(-x));
  }
}

// tx1[ei1[m]] -= ew[m] * x[ei0[m]]      (tx1 pre-initialized with x)
__global__ void k_lap_scatter(const int* __restrict__ ei, const float* __restrict__ ew,
                              const float* __restrict__ x, float* __restrict__ tx1,
                              long long NE, int dv) {
  GSL(i, NE * dv) {
    const long long m = i / dv; const int f = (int)(i - m * dv);
    const int src = ei[m], dst = ei[NE + m];
    atomicAdd(&tx1[(long long)dst * dv + f], -ew[m] * x[(long long)src * dv + f]);
  }
}

__global__ void k_bn_stats(const float* __restrict__ X, long long M, int dv,
                           float* __restrict__ sum, float* __restrict__ sq) {
  GSL(i, M * dv) {
    const int f = (int)(i % dv);
    const float v = X[i];
    atomicAdd(&sum[f], v);
    atomicAdd(&sq[f], v * v);
  }
}

// out[r*ldo + f] = relu(g*(x-mean)*rsqrt(var+eps)+be) ; out already column-offset
__global__ void k_bn_apply_relu(const float* __restrict__ X, long long M, int dv,
                                const float* __restrict__ sum, const float* __restrict__ sq,
                                const float* __restrict__ g, const float* __restrict__ be,
                                float* __restrict__ out, int ldo) {
  GSL(i, M * dv) {
    const long long r = i / dv; const int f = (int)(i - r * dv);
    const float mean = sum[f] / (float)M;
    const float var  = sq[f] / (float)M - mean * mean;
    const float y = g[f] * (X[i] - mean) * rsqrtf(var + 1e-5f) + be[f];
    out[r * (long long)ldo + f] = fmaxf(y, 0.0f);
  }
}

// sums[idx[r]] += X[r, :ncols] ; cnt[idx[r]] += 1
__global__ void k_seg_scatter(const float* __restrict__ X, int ldx,
                              const int* __restrict__ idx, long long M, int ncols,
                              float* __restrict__ sums, int lds, float* __restrict__ cnt) {
  GSL(i, M * ncols) {
    const long long r = i / ncols; const int c = (int)(i - r * ncols);
    const int s = idx[r];
    atomicAdd(&sums[(long long)s * lds + c], X[r * (long long)ldx + c]);
    if (c == 0) atomicAdd(&cnt[s], 1.0f);
  }
}

__global__ void k_seg_div(float* __restrict__ sums, int lds, const float* __restrict__ cnt,
                          long long Nseg, int ncols) {
  GSL(i, Nseg * ncols) {
    const long long s = i / ncols; const int c = (int)(i - s * ncols);
    sums[s * (long long)lds + c] /= fmaxf(cnt[s], 1.0f);
  }
}

// ---------------------------------------------------------------------------

static inline int nblk(long long n) {
  long long b = (n + 255) / 256;
  if (b > 1048576) b = 1048576;
  if (b < 1) b = 1;
  return (int)b;
}

extern "C" void kernel_launch(void* const* d_in, const int* in_sizes, int n_in,
                              void* d_out, int out_size, void* d_ws, size_t ws_size,
                              hipStream_t stream) {
  (void)in_sizes; (void)n_in; (void)out_size; (void)ws_size;

  // ---- problem constants ----
  const int N0 = 38400, E0 = 102400, N1 = 19200, E1 = 51200, B = 128;
  const long long NT0 = 6LL * N0, NS0 = 6LL * E0, NT1 = 6LL * N1, NS1 = 6LL * E1;

  auto F32 = [&](int i) { return (const float*)d_in[i]; };
  auto I32 = [&](int i) { return (const int*)d_in[i]; };

  const float* x_t_feat = F32(0);
  const float* x_s_feat = F32(1);
  const int*   eit0 = I32(2);  const float* ewt0 = F32(3);
  const int*   eis0 = I32(4);  const float* ews0 = F32(5);
  const int*   inc0 = I32(6);
  const int*   pos_t = I32(7); const int* pos_s = I32(8);
  const int*   eit1 = I32(9);  const float* ewt1 = F32(10);
  const int*   eis1 = I32(11); const float* ews1 = F32(12);
  const int*   inc1 = I32(13);
  const int*   n_batch1 = I32(14); const int* s_batch1 = I32(15);

  // ---- params (recursive dict insertion order), base index 16 ----
  const int PB = 16;
  const float* initT_W = F32(PB + 0); const float* initT_b = F32(PB + 1);
  const float* initT_g = F32(PB + 2); const float* initT_be = F32(PB + 3);
  const float* initS_W = F32(PB + 4); const float* initS_b = F32(PB + 5);
  const float* initS_g = F32(PB + 6); const float* initS_be = F32(PB + 7);
  struct BlkP { const float *Wt,*bt,*Ws,*bs,*tW0,*tb,*tW1,*tg,*tbe,*sW0,*sb,*sW1,*sg,*sbe; };
  BlkP blk[6];
  for (int k = 0; k < 6; ++k) {
    const int b0 = PB + 8 + 14 * k;
    blk[k] = { F32(b0+0), F32(b0+1), F32(b0+2),  F32(b0+3),
               F32(b0+4), F32(b0+5), F32(b0+6),  F32(b0+7),  F32(b0+8),
               F32(b0+9), F32(b0+10),F32(b0+11), F32(b0+12), F32(b0+13) };
  }
  // att params (PB+92..95) are dead code in the reference graph: skipped.
  const float* out_W = F32(PB + 96); const float* out_b = F32(PB + 97);

  // ---- workspace bump allocator (byte based, 256B aligned) ----
  char* ws = (char*)d_ws; size_t off = 0;
  auto allocB = [&](size_t bytes) -> void* {
    void* p = (void*)(ws + off);
    off += (bytes + 255) & ~(size_t)255;
    return p;
  };
  auto allocF = [&](long long n) { return (float*)allocB((size_t)n * 4); };

  float* xt0_0 = allocF((long long)N0 * 192);
  float* xs0_0 = allocF((long long)E0 * 192);
  float* xt0_1 = allocF((long long)N1 * 960);
  float* xs0_1 = allocF((long long)E1 * 960);
  float* h_t  = allocF((long long)N1 * 256);
  float* h_s  = allocF((long long)E1 * 256);
  float* agg  = allocF((long long)N1 * 256);
  float* tx1  = allocF((long long)E1 * 256);
  float* ybuf = allocF((long long)E1 * 256);
  float* D0   = allocF(N0);
  float* D1   = allocF(N1);
  float* bn_sum = allocF(256);
  float* bn_sq  = allocF(256);
  float* cnt  = allocF(E1);
  float* cat  = allocF((long long)B * 512);
  float* catc = allocF(B);
  // packed bf16 operand buffers (max A: [E1, 704]; max W^T: [256, 704])
  unsigned short* abf = (unsigned short*)allocB((size_t)E1 * 704 * 2);
  unsigned short* bbf = (unsigned short*)allocB((size_t)256 * 704 * 2);

  auto gemm = [&](const float* A, int lda, const float* W, const float* bias,
                  const float* Cin, float* C, int ldc, int M, int N, int K) {
    const int Kp = (K + 31) & ~31;
    const int Np = (N + 15) & ~15;
    k_a_pack<<<nblk((long long)M * Kp), 256, 0, stream>>>(A, lda, M, K, Kp, abf);
    k_w_pack<<<nblk((long long)Np * Kp), 256, 0, stream>>>(W, N, K, Kp, Np, bbf);
    dim3 g((unsigned)(((M + 15) / 16 + 3) / 4), (unsigned)(Np / 16));
    gemm_bf16_wmma<<<g, dim3(128), 0, stream>>>(abf, bbf, bias, Cin, C, ldc, M, N, Kp);
  };
  auto fill = [&](float* p, float v, long long n) {
    k_fill<<<nblk(n), 256, 0, stream>>>(p, v, n);
  };

  // ---- degrees ----
  fill(D0, 1e-6f, N0);
  k_degree<<<nblk(2LL * E0), 256, 0, stream>>>(inc0, 2LL * E0, D0);
  fill(D1, 1e-6f, N1);
  k_degree<<<nblk(2LL * E1), 256, 0, stream>>>(inc1, 2LL * E1, D1);

  // ---- init linear + BN + ReLU ----
  gemm(x_t_feat, 15, initT_W, initT_b, nullptr, ybuf, 64, N0, 64, 15);
  fill(bn_sum, 0.f, 64); fill(bn_sq, 0.f, 64);
  k_bn_stats<<<nblk((long long)N0 * 64), 256, 0, stream>>>(ybuf, N0, 64, bn_sum, bn_sq);
  k_bn_apply_relu<<<nblk((long long)N0 * 64), 256, 0, stream>>>(
      ybuf, N0, 64, bn_sum, bn_sq, initT_g, initT_be, xt0_0, 192);

  gemm(x_s_feat, 14, initS_W, initS_b, nullptr, ybuf, 64, E0, 64, 14);
  fill(bn_sum, 0.f, 64); fill(bn_sq, 0.f, 64);
  k_bn_stats<<<nblk((long long)E0 * 64), 256, 0, stream>>>(ybuf, E0, 64, bn_sum, bn_sq);
  k_bn_apply_relu<<<nblk((long long)E0 * 64), 256, 0, stream>>>(
      ybuf, E0, 64, bn_sum, bn_sq, initS_g, initS_be, xs0_0, 192);

  // ---- dense blocks ----
  const int FIL[3] = {64, 128, 256};
  int din = 64;
  int wt = 64, wsd = 64;               // current widths of x_t0 / x_s0
  int bidx = 0;

  for (int i = 0; i < 3; ++i) {
    const int dv = FIL[i];
    const int lvl = (i == 0) ? 0 : 1;
    const int N = lvl ? N1 : N0, E = lvl ? E1 : E0, lda = lvl ? 960 : 192;
    float* xt0 = lvl ? xt0_1 : xt0_0;
    float* xs0 = lvl ? xs0_1 : xs0_0;
    const int* eit = lvl ? eit1 : eit0;  const float* ewt = lvl ? ewt1 : ewt0;
    const int* eis = lvl ? eis1 : eis0;  const float* ews = lvl ? ews1 : ews0;
    const int* inc = lvl ? inc1 : inc0;  const float* D = lvl ? D1 : D0;
    const long long NT = lvl ? NT1 : NT0, NS = lvl ? NS1 : NS0;

    for (int j = 0; j < 2; ++j, ++bidx) {
      const BlkP& P = blk[bidx];

      // ne_int: h_t = x_t0 @ Wt + bt ; h_s = x_s0 @ Ws + bs
      gemm(xt0, lda, P.Wt, P.bt, nullptr, h_t, dv, N, dv, din);
      gemm(xs0, lda, P.Ws, P.bs, nullptr, h_s, dv, E, dv, din);
      fill(agg, 0.f, (long long)N * dv);
      k_inc_scatter<<<nblk((long long)E * dv), 256, 0, stream>>>(h_s, inc, E, dv, agg);
      k_edge_sigmoid<<<nblk((long long)E * dv), 256, 0, stream>>>(h_s, h_t, inc, D, E, dv);
      k_node_sigmoid<<<nblk((long long)N * dv), 256, 0, stream>>>(h_t, agg, D, N, dv);

      // hl2 (nodes): y = x@W0 + (x - L x)@W1 + b -> BN -> ReLU -> append
      k_copy<<<nblk((long long)N * dv), 256, 0, stream>>>(tx1, h_t, (long long)N * dv);
      k_lap_scatter<<<nblk(NT * dv), 256, 0, stream>>>(eit, ewt, h_t, tx1, NT, dv);
      gemm(h_t, dv, P.tW0, P.tb, nullptr, ybuf, dv, N, dv, dv);
      gemm(tx1, dv, P.tW1, nullptr, ybuf, ybuf, dv, N, dv, dv);
      fill(bn_sum, 0.f, dv); fill(bn_sq, 0.f, dv);
      k_bn_stats<<<nblk((long long)N * dv), 256, 0, stream>>>(ybuf, N, dv, bn_sum, bn_sq);
      k_bn_apply_relu<<<nblk((long long)N * dv), 256, 0, stream>>>(
          ybuf, N, dv, bn_sum, bn_sq, P.tg, P.tbe, xt0 + wt, lda);

      // hl2 (edges)
      k_copy<<<nblk((long long)E * dv), 256, 0, stream>>>(tx1, h_s, (long long)E * dv);
      k_lap_scatter<<<nblk(NS * dv), 256, 0, stream>>>(eis, ews, h_s, tx1, NS, dv);
      gemm(h_s, dv, P.sW0, P.sb, nullptr, ybuf, dv, E, dv, dv);
      gemm(tx1, dv, P.sW1, nullptr, ybuf, ybuf, dv, E, dv, dv);
      fill(bn_sum, 0.f, dv); fill(bn_sq, 0.f, dv);
      k_bn_stats<<<nblk((long long)E * dv), 256, 0, stream>>>(ybuf, E, dv, bn_sum, bn_sq);
      k_bn_apply_relu<<<nblk((long long)E * dv), 256, 0, stream>>>(
          ybuf, E, dv, bn_sum, bn_sq, P.sg, P.sbe, xs0 + wsd, lda);

      wt += dv; wsd += dv; din += dv;
    }

    if (i == 0) {
      // attention branch is dead code (its outputs are overwritten before use).
      // pool_loc: seg_mean x_t0 / x_s0 -> level-1 (width 192)
      fill(xt0_1, 0.f, (long long)N1 * 960);
      fill(cnt, 0.f, N1);
      k_seg_scatter<<<nblk((long long)N0 * 192), 256, 0, stream>>>(
          xt0_0, 192, pos_t, N0, 192, xt0_1, 960, cnt);
      k_seg_div<<<nblk((long long)N1 * 192), 256, 0, stream>>>(xt0_1, 960, cnt, N1, 192);

      fill(xs0_1, 0.f, (long long)E1 * 960);
      fill(cnt, 0.f, E1);
      k_seg_scatter<<<nblk((long long)E0 * 192), 256, 0, stream>>>(
          xs0_0, 192, pos_s, E0, 192, xs0_1, 960, cnt);
      k_seg_div<<<nblk((long long)E1 * 192), 256, 0, stream>>>(xs0_1, 960, cnt, E1, 192);
    }
  }

  // ---- readout: concat(seg_mean(x_s, s_batch1), seg_mean(x_t, n_batch1)) @ W + b
  // last block outputs live in columns [704, 960) of the level-1 dense buffers.
  fill(cat, 0.f, (long long)B * 512);
  fill(catc, 0.f, B);
  k_seg_scatter<<<nblk((long long)E1 * 256), 256, 0, stream>>>(
      xs0_1 + 704, 960, s_batch1, E1, 256, cat, 512, catc);
  k_seg_div<<<nblk((long long)B * 256), 256, 0, stream>>>(cat, 512, catc, B, 256);
  fill(catc, 0.f, B);
  k_seg_scatter<<<nblk((long long)N1 * 256), 256, 0, stream>>>(
      xt0_1 + 704, 960, n_batch1, N1, 256, cat + 256, 512, catc);
  k_seg_div<<<nblk((long long)B * 256), 256, 0, stream>>>(cat + 256, 512, catc, B, 256);

  gemm(cat, 512, out_W, out_b, nullptr, (float*)d_out, 10, B, 10, 512);
}
